// LSTM_42786464202960
// MI455X (gfx1250) — matrix operand order repaired and checked by reading, not compiled, not run on hardware
//
#include <hip/hip_runtime.h>
#include <hip/hip_bf16.h>

typedef __attribute__((ext_vector_type(16))) __bf16 v16bf;
typedef __attribute__((ext_vector_type(8)))  __bf16 v8bf;
typedef __attribute__((ext_vector_type(8)))  float  v8f;
typedef __attribute__((ext_vector_type(4)))  float  v4f;

#define B_  4096
#define T_  128
#define I_  128
#define H_  32
#define O_  128
#define WAVES_PER_BLOCK 2

// Fast sigmoid/tanh: v_exp_f32 + v_rcp_f32 (avoid the IEEE-precise
// v_div_scale/div_fmas/div_fixup sequence the compiler emits for '/').
__device__ __forceinline__ float fsig(float x) {
    return __builtin_amdgcn_rcpf(1.0f + __expf(-x));
}
__device__ __forceinline__ float ftanhf(float x) {
    return __builtin_fmaf(2.0f, __builtin_amdgcn_rcpf(1.0f + __expf(-2.0f * x)), -1.0f);
}

__device__ __forceinline__ v8f vzero8() {
    v8f a;
#pragma unroll
    for (int i = 0; i < 8; ++i) a[i] = 0.0f;
    return a;
}
__device__ __forceinline__ v16bf bzero16() {
    v16bf a;
#pragma unroll
    for (int i = 0; i < 16; ++i) a[i] = (__bf16)0.0f;
    return a;
}

// Fused 2-layer LSTM + last-step linear.
// One wave owns 16 batch rows for the whole T loop.
// Weight B-fragments (bf16) live in LDS; h D->A transposes go through a
// per-wave LDS staging tile (wave-internal, in-order DS, no barrier).
__global__ __launch_bounds__(32 * WAVES_PER_BLOCK, 1)
void lstm_fused(const float* __restrict__ x,
                const float* __restrict__ W_ih0, const float* __restrict__ W_hh0,
                const float* __restrict__ b_ih0, const float* __restrict__ b_hh0,
                const float* __restrict__ W_ih1, const float* __restrict__ W_hh1,
                const float* __restrict__ b_ih1, const float* __restrict__ b_hh1,
                const float* __restrict__ W_lin, const float* __restrict__ b_lin,
                float* __restrict__ out)
{
    // 64 B-fragments * 512 bf16 each:
    //  [0,32): W_ih0^T  (kc*8 + nt)   [32,40): W_hh0^T   [40,48): W_ih1^T
    //  [48,56): W_hh1^T               [56,64): W_lin^T
    __shared__ __attribute__((aligned(32))) __bf16 wlds[64 * 512];
    __shared__ __attribute__((aligned(32))) __bf16 hstage[WAVES_PER_BLOCK][2][16 * 32];

    const int tid  = threadIdx.x;
    const int lane = tid & 31;
    const int wave = tid >> 5;
    const int lmod = lane & 15;   // N (or M) within 16
    const int lhi  = lane >> 4;   // lane half-group

    // ---- one-time: convert weights fp32->bf16 into B-fragment layout ----
    // B frag (K=32 x N=16): lane holds N = lane%16, e0..7 -> K=(lane/16)*8+e, e8..15 -> K=16+(lane/16)*8+(e-8).
    for (int idx = tid; idx < 64 * 512; idx += 32 * WAVES_PER_BLOCK) {
        const int frag = idx >> 9;
        const int rem  = idx & 511;
        const int l    = rem >> 4;
        const int e    = rem & 15;
        const int n16  = l & 15;
        const int kl   = ((e >> 3) << 4) + ((l >> 4) << 3) + (e & 7);
        float v;
        if (frag < 32) {            // W_ih0 [128 gates x 128 in], K = kc*32 + kl
            const int kc = frag >> 3, nt = frag & 7;
            v = W_ih0[(nt * 16 + n16) * I_ + kc * 32 + kl];
        } else if (frag < 40) {     // W_hh0 [128 x 32]
            const int nt = frag - 32;
            v = W_hh0[(nt * 16 + n16) * H_ + kl];
        } else if (frag < 48) {     // W_ih1 [128 x 32]
            const int nt = frag - 40;
            v = W_ih1[(nt * 16 + n16) * H_ + kl];
        } else if (frag < 56) {     // W_hh1 [128 x 32]
            const int nt = frag - 48;
            v = W_hh1[(nt * 16 + n16) * H_ + kl];
        } else {                    // W_lin [128 out x 32]
            const int nt = frag - 56;
            v = W_lin[(nt * 16 + n16) * H_ + kl];
        }
        wlds[idx] = (__bf16)v;
    }
    __syncthreads();

    // per-lane gate biases (column-broadcast in C/D layout: N = lane%16)
    float bias0[8], bias1[8], biasL[8];
#pragma unroll
    for (int n = 0; n < 8; ++n) {
        bias0[n] = b_ih0[n * 16 + lmod] + b_hh0[n * 16 + lmod];
        bias1[n] = b_ih1[n * 16 + lmod] + b_hh1[n * 16 + lmod];
        biasL[n] = b_lin[n * 16 + lmod];
    }

    const int b0 = (blockIdx.x * WAVES_PER_BLOCK + wave) * 16;
    __bf16* stage1 = &hstage[wave][0][0];
    __bf16* stage2 = &hstage[wave][1][0];

    // load one 32-byte B fragment from LDS
    auto loadB = [&](int frag) -> v16bf {
        return *(const v16bf*)&wlds[frag * 512 + lane * 16];
    };
    // A frag (16xK=32): lane M = lane%16; e0..7 -> K=lhi*8+0..7, e8..15 -> K=16+lhi*8+0..7
    auto loadA = [&](const __bf16* st) -> v16bf {
        const __bf16* p = st + lmod * 32 + lhi * 8;
        v8bf lo = *(const v8bf*)p;
        v8bf hi = *(const v8bf*)(p + 16);
        v16bf r;
#pragma unroll
        for (int e = 0; e < 8; ++e) { r[e] = lo[e]; r[8 + e] = hi[e]; }
        return r;
    };
    // D frags (hidden cols 0..15 and 16..31) -> staging tile [M][k] bf16
    auto storeH = [&](__bf16* st, const v8f& d0, const v8f& d1) {
#pragma unroll
        for (int r = 0; r < 8; ++r) {
            const int M = r + lhi * 8;
            st[M * 32 + lmod]      = (__bf16)d0[r];
            st[M * 32 + 16 + lmod] = (__bf16)d1[r];
        }
    };
    // LSTM cell: gates in 8 D frags (i:0-1, f:2-3, g:4-5, o:6-7), c kept as 2 D frags
    auto cell = [&](v8f* g, v8f& ca, v8f& cb, v8f& hd0, v8f& hd1) {
#pragma unroll
        for (int r = 0; r < 8; ++r) {
            float i0 = fsig(g[0][r]), f0 = fsig(g[2][r]);
            float g0 = ftanhf(g[4][r]), o0 = fsig(g[6][r]);
            float c0 = __builtin_fmaf(f0, ca[r], i0 * g0);
            ca[r] = c0; hd0[r] = o0 * ftanhf(c0);
            float i1 = fsig(g[1][r]), f1 = fsig(g[3][r]);
            float g1 = ftanhf(g[5][r]), o1 = fsig(g[7][r]);
            float c1 = __builtin_fmaf(f1, cb[r], i1 * g1);
            cb[r] = c1; hd1[r] = o1 * ftanhf(c1);
        }
    };

    v16bf h1A = bzero16(), h2A = bzero16();
    v8f c1a = vzero8(), c1b = vzero8(), c2a = vzero8(), c2b = vzero8();

    const float* xp = x + (size_t)(b0 + lmod) * T_ * I_;  // lane's batch row

    for (int t = 0; t < T_; ++t, xp += I_) {
        // ---- x_t -> 4 bf16 A fragments (K chunks of 32) ----
        v16bf xA[4];
#pragma unroll
        for (int kc = 0; kc < 4; ++kc) {
            const float* p = xp + kc * 32 + lhi * 8;
            v4f a0 = *(const v4f*)p;
            v4f a1 = *(const v4f*)(p + 4);
            v4f b0v = *(const v4f*)(p + 16);
            v4f b1v = *(const v4f*)(p + 20);
            v16bf f;
#pragma unroll
            for (int e = 0; e < 4; ++e) {
                f[e]      = (__bf16)a0[e];
                f[e + 4]  = (__bf16)a1[e];
                f[e + 8]  = (__bf16)b0v[e];
                f[e + 12] = (__bf16)b1v[e];
            }
            xA[kc] = f;
        }
        if (t + 1 < T_) {
#pragma unroll
            for (int kc = 0; kc < 4; ++kc)
                __builtin_prefetch(xp + I_ + kc * 32 + lhi * 16, 0, 3);
        }

        // ---- layer 1 gates: x@W_ih0^T + h1@W_hh0^T + b ----
        v8f acc[8];
#pragma unroll
        for (int n = 0; n < 8; ++n) {
            v8f a = vzero8();
#pragma unroll
            for (int kc = 0; kc < 4; ++kc)
                a = __builtin_amdgcn_wmma_f32_16x16x32_bf16(
                        false, xA[kc], false, loadB(kc * 8 + n), (short)0, a, false, false);
            a = __builtin_amdgcn_wmma_f32_16x16x32_bf16(
                    false, h1A, false, loadB(32 + n), (short)0, a, false, false);
#pragma unroll
            for (int r = 0; r < 8; ++r) a[r] += bias0[n];
            acc[n] = a;
        }
        v8f h1d0, h1d1;
        cell(acc, c1a, c1b, h1d0, h1d1);
        storeH(stage1, h1d0, h1d1);      // D layout -> LDS
        h1A = loadA(stage1);             // -> A layout (wave-internal transpose)

        // ---- layer 2 gates: h1@W_ih1^T + h2@W_hh1^T + b ----
        v8f acc2[8];
#pragma unroll
        for (int n = 0; n < 8; ++n) {
            v8f a = vzero8();
            a = __builtin_amdgcn_wmma_f32_16x16x32_bf16(
                    false, h1A, false, loadB(40 + n), (short)0, a, false, false);
            a = __builtin_amdgcn_wmma_f32_16x16x32_bf16(
                    false, h2A, false, loadB(48 + n), (short)0, a, false, false);
#pragma unroll
            for (int r = 0; r < 8; ++r) a[r] += bias1[n];
            acc2[n] = a;
        }
        v8f h2d0, h2d1;
        cell(acc2, c2a, c2b, h2d0, h2d1);
        storeH(stage2, h2d0, h2d1);
        h2A = loadA(stage2);
    }

    // ---- final linear on h2[T-1]: out[b,o] = h2 @ W_lin^T + b_lin ----
#pragma unroll
    for (int n = 0; n < 8; ++n) {
        v8f a = __builtin_amdgcn_wmma_f32_16x16x32_bf16(
                    false, h2A, false, loadB(56 + n), (short)0, vzero8(), false, false);
#pragma unroll
        for (int r = 0; r < 8; ++r) {
            const int b = b0 + r + lhi * 8;
            out[(size_t)b * O_ + n * 16 + lmod] = a[r] + biasL[n];
        }
    }
}

extern "C" void kernel_launch(void* const* d_in, const int* in_sizes, int n_in,
                              void* d_out, int out_size, void* d_ws, size_t ws_size,
                              hipStream_t stream) {
    const float* x     = (const float*)d_in[0];
    const float* W_ih0 = (const float*)d_in[1];
    const float* W_hh0 = (const float*)d_in[2];
    const float* b_ih0 = (const float*)d_in[3];
    const float* b_hh0 = (const float*)d_in[4];
    const float* W_ih1 = (const float*)d_in[5];
    const float* W_hh1 = (const float*)d_in[6];
    const float* b_ih1 = (const float*)d_in[7];
    const float* b_hh1 = (const float*)d_in[8];
    const float* W_lin = (const float*)d_in[9];
    const float* b_lin = (const float*)d_in[10];
    // d_in[11] = seq_len (compile-time T_ = 128)
    float* out = (float*)d_out;

    const int tiles  = B_ / 16;                       // 256 waves
    const int blocks = tiles / WAVES_PER_BLOCK;       // 128 blocks x 64 threads
    lstm_fused<<<blocks, 32 * WAVES_PER_BLOCK, 0, stream>>>(
        x, W_ih0, W_hh0, b_ih0, b_hh0,
        W_ih1, W_hh1, b_ih1, b_hh1, W_lin, b_lin, out);
}
